// UpdateBlock_66803921322275
// MI455X (gfx1250) — compile-verified, matrix operand order checked by reading, not run on hardware
//
#include <hip/hip_runtime.h>
#include <cstdint>

#define DEV static __device__ __forceinline__

typedef __attribute__((ext_vector_type(16))) __bf16 v16bf;
typedef __attribute__((ext_vector_type(8)))  __bf16 v8bf;
typedef __attribute__((ext_vector_type(8)))  float  v8f;

constexpr int NPTS = 8192;   // points per batch
constexpr int BB   = 2;      // batch
constexpr int DD   = 64;     // feature dim
constexpr int KNBR = 16;     // graph neighbors

// ---------------------------------------------------------------------------
// WMMA helpers (CDNA5 wave32, 16x16x32 bf16 -> f32)
// A-matrix / B-matrix per-lane layout is identical when the source tile is
// row-major with 32 contiguous "K" elements per row chunk:
//   lane = {row = lane&15, hi = lane>>4}
//   elems[0..7]  = row[chunk + hi*8      .. +7]
//   elems[8..15] = row[chunk + 16 + hi*8 .. +7]
// ---------------------------------------------------------------------------
DEV v16bf load_frag(const __bf16* base, int stride, int lane, int chunk) {
  const int row = lane & 15;
  const int hi  = lane >> 4;
  const __bf16* p = base + (size_t)row * stride + chunk + hi * 8;
  v8bf lo = *reinterpret_cast<const v8bf*>(p);
  v8bf hv = *reinterpret_cast<const v8bf*>(p + 16);
  return __builtin_shufflevector(lo, hv, 0,1,2,3,4,5,6,7,8,9,10,11,12,13,14,15);
}

DEV v8f wmma_bf16(v16bf a, v16bf b, v8f c) {
  return __builtin_amdgcn_wmma_f32_16x16x32_bf16(false, a, false, b,
                                                 (short)0, c, false, false);
}

DEV float sigmoidf_(float x) { return 1.0f / (1.0f + __expf(-x)); }

// Online-softmax step for one 16-query group over 32 keys held in st0/st1
// (S^T C-layout). Updates m/lsum, rescales the 4 accumulator tiles, and
// returns the probability B-operand (pure per-lane remap, no shuffles).
DEV v16bf softmax_step(v8f st0, v8f st1, float& m, float& lsum, v8f acc[4]) {
  float tmax = -INFINITY;
  #pragma unroll
  for (int i = 0; i < 8; ++i) tmax = fmaxf(tmax, fmaxf(st0[i], st1[i]));
  tmax = fmaxf(tmax, __shfl_xor(tmax, 16, 32));
  const float mnew = fmaxf(m, tmax);
  const float cs   = __expf(m - mnew);

  float p0[8], p1[8], ps = 0.0f;
  #pragma unroll
  for (int i = 0; i < 8; ++i) {
    p0[i] = __expf(st0[i] - mnew);
    p1[i] = __expf(st1[i] - mnew);
    ps += p0[i] + p1[i];
  }
  ps += __shfl_xor(ps, 16, 32);
  lsum = lsum * cs + ps;
  m = mnew;

  #pragma unroll
  for (int dt = 0; dt < 4; ++dt) {
    #pragma unroll
    for (int i = 0; i < 8; ++i) acc[dt][i] *= cs;
  }

  v16bf pb;
  #pragma unroll
  for (int i = 0; i < 8; ++i) { pb[i] = (__bf16)p0[i]; pb[i + 8] = (__bf16)p1[i]; }
  return pb;
}

// ---------------------------------------------------------------------------
// Kernel 1: MotionEncoder + q/k/v/pos projections. One thread per point.
//   motion (b,64,n) fp32 | q (b,n,64) bf16 (x 1/sqrt(64)) | k (b,n,64) bf16
//   v (b,64,n) bf16 dim-major (PV A-operand layout)
// ---------------------------------------------------------------------------
__global__ __launch_bounds__(256) void encoder_kernel(
    const float* __restrict__ inp,  const float* __restrict__ corr,
    const float* __restrict__ flow, const float* __restrict__ xyz,
    const float* __restrict__ w_mc, const float* __restrict__ b_mc,
    const float* __restrict__ w_mf, const float* __restrict__ b_mf,
    const float* __restrict__ w_me, const float* __restrict__ b_me,
    const float* __restrict__ w_qg, const float* __restrict__ b_qg,
    const float* __restrict__ w_kg, const float* __restrict__ b_kg,
    const float* __restrict__ w_vg, const float* __restrict__ b_vg,
    const float* __restrict__ w_pos,const float* __restrict__ b_pos,
    float* __restrict__ motion_out,
    __bf16* __restrict__ qb, __bf16* __restrict__ kb, __bf16* __restrict__ vb)
{
  const int t = blockIdx.x * blockDim.x + threadIdx.x;
  if (t >= BB * NPTS) return;
  const int b = t / NPTS, n = t % NPTS;

  const float* cc = corr + (size_t)b * DD * NPTS + n;
  const float* ic = inp  + (size_t)b * DD * NPTS + n;

  float f3[3], x3[3];
  for (int j = 0; j < 3; ++j) {
    f3[j] = flow[((size_t)b * NPTS + n) * 3 + j];
    x3[j] = xyz [((size_t)b * NPTS + n) * 3 + j];
  }

  float cv[DD], iv[DD];
  for (int c = 0; c < DD; ++c) { cv[c] = cc[(size_t)c * NPTS]; iv[c] = ic[(size_t)c * NPTS]; }

  float corf[DD], flo[DD];
  for (int o = 0; o < DD; ++o) {
    float a = b_mc[o];
    for (int c = 0; c < DD; ++c) a += w_mc[o * DD + c] * cv[c];
    corf[o] = fmaxf(a, 0.0f);
    float f = b_mf[o];
    for (int j = 0; j < 3; ++j) f += w_mf[o * 3 + j] * f3[j];
    flo[o] = fmaxf(f, 0.0f);
  }

  float mot[DD];
  for (int o = 0; o < 61; ++o) {
    float a = b_me[o];
    for (int c = 0; c < DD; ++c) a += w_me[o * 128 + c]      * corf[c];
    for (int c = 0; c < DD; ++c) a += w_me[o * 128 + 64 + c] * flo[c];
    mot[o] = fmaxf(a, 0.0f);
  }
  mot[61] = f3[0]; mot[62] = f3[1]; mot[63] = f3[2];

  float* mo = motion_out + (size_t)b * DD * NPTS + n;
  for (int c = 0; c < DD; ++c) mo[(size_t)c * NPTS] = mot[c];

  __bf16* qp = qb + ((size_t)b * NPTS + n) * DD;
  __bf16* kp = kb + ((size_t)b * NPTS + n) * DD;
  __bf16* vp = vb + (size_t)b * DD * NPTS + n;
  for (int o = 0; o < DD; ++o) {
    float pos = b_pos[o];
    for (int j = 0; j < 3; ++j) pos += w_pos[o * 3 + j] * x3[j];
    float q = b_qg[o] + pos, k = b_kg[o] + pos, v = b_vg[o];
    for (int c = 0; c < DD; ++c) {
      q += w_qg[o * DD + c] * iv[c];
      k += w_kg[o * DD + c] * iv[c];
      v += w_vg[o * DD + c] * mot[c];
    }
    qp[o] = (__bf16)(q * 0.125f);
    kp[o] = (__bf16)k;
    vp[(size_t)o * NPTS] = (__bf16)v;
  }
}

// ---------------------------------------------------------------------------
// Kernel 2: flash-attention with WMMA. One wave owns 32 queries (two 16-query
// groups A/B sharing every K/V fragment -> 32 flop/byte from L2) and streams
// all 8192 keys in 32-key steps with online softmax.
// Per step: 8 score WMMAs + 8 PV WMMAs. All K fragments are loaded before the
// first WMMA so in-order load completion allows partial loadcnt waits; V
// fragments are issued before the softmax math so their latency hides under
// the exp chain.
// ---------------------------------------------------------------------------
__global__ __launch_bounds__(32) void attn_kernel(
    const __bf16* __restrict__ qb, const __bf16* __restrict__ kb,
    const __bf16* __restrict__ vb, const float* __restrict__ gamma_p,
    float* __restrict__ gma)
{
  const int wave = blockIdx.x;              // 0 .. BB*NPTS/32-1
  const int b    = wave / (NPTS / 32);
  const int qt   = wave % (NPTS / 32);      // 32-query tile index
  const int lane = threadIdx.x;
  const int hi   = lane >> 4;

  const __bf16* QbaseA = qb + ((size_t)b * NPTS + (size_t)qt * 32) * DD;
  const __bf16* QbaseB = QbaseA + (size_t)16 * DD;
  const __bf16* Kbase  = kb + (size_t)b * NPTS * DD;
  const __bf16* Vbase  = vb + (size_t)b * DD * NPTS;

  const v16bf qA0 = load_frag(QbaseA, DD, lane, 0);
  const v16bf qA1 = load_frag(QbaseA, DD, lane, 32);
  const v16bf qB0 = load_frag(QbaseB, DD, lane, 0);
  const v16bf qB1 = load_frag(QbaseB, DD, lane, 32);

  v8f accA[4], accB[4];
  #pragma unroll
  for (int dt = 0; dt < 4; ++dt) { accA[dt] = v8f{}; accB[dt] = v8f{}; }
  float mA = -INFINITY, lA = 0.0f;
  float mB = -INFINITY, lB = 0.0f;

  for (int j0 = 0; j0 < NPTS; j0 += 32) {
    if (j0 + 32 < NPTS) {
      __builtin_prefetch(Kbase + (size_t)(j0 + 32) * DD + lane * 64, 0, 1);
      __builtin_prefetch(Vbase + (size_t)lane * NPTS + (j0 + 32), 0, 1);
    }

    // ---- all K fragments for this step, before any WMMA ----
    const __bf16* K0 = Kbase + (size_t)j0 * DD;
    const __bf16* K1 = Kbase + (size_t)(j0 + 16) * DD;
    const v16bf k00 = load_frag(K0, DD, lane, 0);
    const v16bf k01 = load_frag(K0, DD, lane, 32);
    const v16bf k10 = load_frag(K1, DD, lane, 0);
    const v16bf k11 = load_frag(K1, DD, lane, 32);

    // ---- V fragments issued early; consumed only after the softmax ----
    v16bf va[4];
    #pragma unroll
    for (int dt = 0; dt < 4; ++dt)
      va[dt] = load_frag(Vbase + (size_t)(16 * dt) * NPTS + j0, NPTS, lane, 0);

    // ---- scores S^T: keys x queries, both query groups reuse K frags ----
    v8f stA0 = v8f{}, stA1 = v8f{}, stB0 = v8f{}, stB1 = v8f{};
    stA0 = wmma_bf16(k00, qA0, stA0);
    stA0 = wmma_bf16(k01, qA1, stA0);
    stB0 = wmma_bf16(k00, qB0, stB0);
    stB0 = wmma_bf16(k01, qB1, stB0);
    stA1 = wmma_bf16(k10, qA0, stA1);
    stA1 = wmma_bf16(k11, qA1, stA1);
    stB1 = wmma_bf16(k10, qB0, stB1);
    stB1 = wmma_bf16(k11, qB1, stB1);

    // ---- online softmax per group ----
    const v16bf pbA = softmax_step(stA0, stA1, mA, lA, accA);
    const v16bf pbB = softmax_step(stB0, stB1, mB, lB, accB);

    // ---- O += V x P, V frags reused by both groups ----
    #pragma unroll
    for (int dt = 0; dt < 4; ++dt) {
      accA[dt] = wmma_bf16(va[dt], pbA, accA[dt]);
      accB[dt] = wmma_bf16(va[dt], pbB, accB[dt]);
    }
  }

  const float g    = gamma_p[0];
  const float invA = g / lA;
  const float invB = g / lB;
  float* ogA = gma + (size_t)b * DD * NPTS + (size_t)qt * 32 + (lane & 15);
  float* ogB = ogA + 16;
  #pragma unroll
  for (int dt = 0; dt < 4; ++dt) {
    #pragma unroll
    for (int r = 0; r < 8; ++r) {
      ogA[(size_t)(16 * dt + r + 8 * hi) * NPTS] = accA[dt][r] * invA;
      ogB[(size_t)(16 * dt + r + 8 * hi) * NPTS] = accB[dt][r] * invB;
    }
  }
}

// ---------------------------------------------------------------------------
// Kernel 3: ConvGRU. hx = [net, inp, motion, gma] (256 ch).
// ---------------------------------------------------------------------------
DEV float dot256(const float* __restrict__ w, const float* a, const float* b2,
                 const float* c, const float* d) {
  float s = 0.0f;
  for (int i = 0; i < 64; ++i) s += w[i]       * a[i];
  for (int i = 0; i < 64; ++i) s += w[64 + i]  * b2[i];
  for (int i = 0; i < 64; ++i) s += w[128 + i] * c[i];
  for (int i = 0; i < 64; ++i) s += w[192 + i] * d[i];
  return s;
}

__global__ __launch_bounds__(256) void gru_kernel(
    const float* __restrict__ net, const float* __restrict__ inp,
    const float* __restrict__ motion, const float* __restrict__ gma,
    const float* __restrict__ w_gz, const float* __restrict__ b_gz,
    const float* __restrict__ w_gr, const float* __restrict__ b_gr,
    const float* __restrict__ w_gq, const float* __restrict__ b_gq,
    float* __restrict__ net_out)
{
  const int t = blockIdx.x * blockDim.x + threadIdx.x;
  if (t >= BB * NPTS) return;
  const int b = t / NPTS, n = t % NPTS;

  float nv[DD], iv[DD], mv[DD], gv[DD];
  for (int c = 0; c < DD; ++c) {
    nv[c] = net   [(size_t)b * DD * NPTS + (size_t)c * NPTS + n];
    iv[c] = inp   [(size_t)b * DD * NPTS + (size_t)c * NPTS + n];
    mv[c] = motion[(size_t)b * DD * NPTS + (size_t)c * NPTS + n];
    gv[c] = gma   [(size_t)b * DD * NPTS + (size_t)c * NPTS + n];
  }

  float rn[DD];
  for (int o = 0; o < DD; ++o) {
    float r = sigmoidf_(b_gr[o] + dot256(w_gr + o * 256, nv, iv, mv, gv));
    rn[o] = r * nv[o];
  }
  for (int o = 0; o < DD; ++o) {
    float z = sigmoidf_(b_gz[o] + dot256(w_gz + o * 256, nv, iv, mv, gv));
    float q = tanhf(b_gq[o] + dot256(w_gq + o * 256, rn, iv, mv, gv));
    net_out[(size_t)b * DD * NPTS + (size_t)o * NPTS + n] =
        (1.0f - z) * nv[o] + z * q;
  }
}

// ---------------------------------------------------------------------------
// Kernel 4: FlowHead + graph SetConv (gather 16 neighbors, relu-max pool).
// ---------------------------------------------------------------------------
__global__ __launch_bounds__(256) void flowhead_kernel(
    const float* __restrict__ net_out, const int* __restrict__ graph,
    const float* __restrict__ w_f1, const float* __restrict__ b_f1,
    const float* __restrict__ w_set, const float* __restrict__ b_set,
    const float* __restrict__ w_o1, const float* __restrict__ b_o1,
    const float* __restrict__ w_o2, const float* __restrict__ b_o2,
    float* __restrict__ dflow)
{
  const int t = blockIdx.x * blockDim.x + threadIdx.x;
  if (t >= BB * NPTS) return;
  const int b = t / NPTS, n = t % NPTS;
  const float* nb = net_out + (size_t)b * DD * NPTS;

  float nv[DD];
  for (int c = 0; c < DD; ++c) nv[c] = nb[(size_t)c * NPTS + n];

  float o1[DD];
  for (int o = 0; o < DD; ++o) {
    float a = b_f1[o];
    for (int c = 0; c < DD; ++c) a += w_f1[o * DD + c] * nv[c];
    o1[o] = a;                          // no relu on o1 in reference
  }

  float s[DD];
  for (int o = 0; o < DD; ++o) s[o] = 0.0f;   // relu(..) >= 0 so 0 is safe
  for (int k = 0; k < KNBR; ++k) {
    const int idx = graph[((size_t)b * NPTS + n) * KNBR + k];
    float fv[DD];
    for (int c = 0; c < DD; ++c) fv[c] = nb[(size_t)c * NPTS + idx];
    for (int o = 0; o < DD; ++o) {
      float a = b_set[o];
      for (int c = 0; c < DD; ++c) a += w_set[o * DD + c] * fv[c];
      s[o] = fmaxf(s[o], fmaxf(a, 0.0f));
    }
  }

  float u[DD];
  for (int o = 0; o < DD; ++o) {
    float a = b_o1[o];
    for (int c = 0; c < DD; ++c) a += w_o1[o * 128 + c]      * s[c];
    for (int c = 0; c < DD; ++c) a += w_o1[o * 128 + 64 + c] * o1[c];
    u[o] = fmaxf(a, 0.0f);
  }
  for (int j = 0; j < 3; ++j) {
    float a = b_o2[j];
    for (int c = 0; c < DD; ++c) a += w_o2[j * DD + c] * u[c];
    dflow[((size_t)b * NPTS + n) * 3 + j] = a;
  }
}

// ---------------------------------------------------------------------------
extern "C" void kernel_launch(void* const* d_in, const int* in_sizes, int n_in,
                              void* d_out, int out_size, void* d_ws, size_t ws_size,
                              hipStream_t stream) {
  (void)in_sizes; (void)n_in; (void)out_size; (void)ws_size;

  const float* net   = (const float*)d_in[0];
  const float* inp   = (const float*)d_in[1];
  const float* corr  = (const float*)d_in[2];
  const float* flow  = (const float*)d_in[3];
  const float* xyz1  = (const float*)d_in[4];
  const int*   graph = (const int*)  d_in[5];
  const float *w_mc = (const float*)d_in[6],  *b_mc = (const float*)d_in[7];
  const float *w_mf = (const float*)d_in[8],  *b_mf = (const float*)d_in[9];
  const float *w_me = (const float*)d_in[10], *b_me = (const float*)d_in[11];
  const float *w_qg = (const float*)d_in[12], *b_qg = (const float*)d_in[13];
  const float *w_kg = (const float*)d_in[14], *b_kg = (const float*)d_in[15];
  const float *w_vg = (const float*)d_in[16], *b_vg = (const float*)d_in[17];
  const float *w_pos= (const float*)d_in[18], *b_pos= (const float*)d_in[19];
  const float *gamma= (const float*)d_in[20];
  const float *w_gz = (const float*)d_in[21], *b_gz = (const float*)d_in[22];
  const float *w_gr = (const float*)d_in[23], *b_gr = (const float*)d_in[24];
  const float *w_gq = (const float*)d_in[25], *b_gq = (const float*)d_in[26];
  const float *w_f1 = (const float*)d_in[27], *b_f1 = (const float*)d_in[28];
  const float *w_set= (const float*)d_in[29], *b_set= (const float*)d_in[30];
  const float *w_o1 = (const float*)d_in[31], *b_o1 = (const float*)d_in[32];
  const float *w_o2 = (const float*)d_in[33], *b_o2 = (const float*)d_in[34];

  float* net_out = (float*)d_out;                       // (B,64,N)
  float* dflow   = net_out + (size_t)BB * DD * NPTS;    // (B,N,3)

  // workspace layout (~14 MB)
  float*  motion = (float*)d_ws;                              // 4 MB
  float*  gma    = motion + (size_t)BB * DD * NPTS;           // 4 MB
  __bf16* qb     = (__bf16*)(gma + (size_t)BB * DD * NPTS);   // 2 MB
  __bf16* kb     = qb + (size_t)BB * NPTS * DD;               // 2 MB
  __bf16* vb     = kb + (size_t)BB * NPTS * DD;               // 2 MB

  const dim3 blk(256);
  const dim3 grd((BB * NPTS + 255) / 256);

  encoder_kernel<<<grd, blk, 0, stream>>>(
      inp, corr, flow, xyz1, w_mc, b_mc, w_mf, b_mf, w_me, b_me,
      w_qg, b_qg, w_kg, b_kg, w_vg, b_vg, w_pos, b_pos,
      motion, qb, kb, vb);

  attn_kernel<<<dim3(BB * NPTS / 32), dim3(32), 0, stream>>>(
      qb, kb, vb, gamma, gma);

  gru_kernel<<<grd, blk, 0, stream>>>(
      net, inp, motion, gma, w_gz, b_gz, w_gr, b_gr, w_gq, b_gq, net_out);

  flowhead_kernel<<<grd, blk, 0, stream>>>(
      net_out, graph, w_f1, b_f1, w_set, b_set, w_o1, b_o1, w_o2, b_o2, dflow);
}
